// FusedSDPAWithKVCache_45440753992394
// MI455X (gfx1250) — matrix operand order repaired and checked by reading, not compiled, not run on hardware
//
#include <hip/hip_runtime.h>
#include <cstdint>
#include <cstddef>

// Problem dims (match reference)
constexpr int Bn = 4;      // batch
constexpr int Cc = 4096;   // HID (channels)
constexpr int Sn = 2048;   // sequence
constexpr int Hn = 32;     // heads
constexpr int Dn = 128;    // head dim

typedef __attribute__((ext_vector_type(16))) __bf16 v16bf;
typedef __attribute__((ext_vector_type(8)))  float  v8f;
typedef int vint4 __attribute__((vector_size(16)));   // async builtin's pointee type

// TDM descriptor group types (probe-verified arity; vector_size style)
typedef unsigned int tv4u __attribute__((vector_size(16)));
typedef int          tv4i __attribute__((vector_size(16)));
typedef int          tv8i __attribute__((vector_size(32)));

using u16 = uint16_t;

#if __has_builtin(__builtin_amdgcn_global_load_async_to_lds_b128) && \
    __has_builtin(__builtin_amdgcn_s_wait_asynccnt)
#define USE_ASYNC_LDS 1
#else
#define USE_ASYNC_LDS 0
#endif

#if __has_builtin(__builtin_amdgcn_tensor_load_to_lds) && \
    __has_builtin(__builtin_amdgcn_s_wait_tensorcnt)
#define USE_TDM 1
#else
#define USE_TDM 0
#endif

// ---------------------------------------------------------------- helpers

__device__ __forceinline__ u16 f2bf_bits(float f) {
    union { float f; uint32_t u; } v; v.f = f;
    uint32_t r = (v.u + 0x7FFFu + ((v.u >> 16) & 1u)) >> 16;   // RNE
    return (u16)r;
}

union FragU { v16bf v; uint4 q[2]; };
union Pack8 { u16 u[8]; uint4 q; };

#if USE_ASYNC_LDS
// one 16-byte async global->LDS copy (tracked by ASYNCcnt)
__device__ __forceinline__ void async_ld16(const u16* g, u16* l) {
    __builtin_amdgcn_global_load_async_to_lds_b128(
        (vint4*)(void*)(const_cast<u16*>(g)),
        (__attribute__((address_space(3))) vint4*)(void*)l,
        0, 0);
}
#endif

#if USE_TDM
// TDM: DMA a 2D bf16 tile (tile_dim0 x tile_dim1, row stride in elements)
// from global into LDS, padding each 16-DWORD row with 4 DWORDs (-> [rows][40]
// u16 LDS layout).  ISA 8.3/8.4 D# packing; 2D tensor -> groups 2/3 zero.
__device__ __forceinline__ void tdm_load_tile_2d(const u16* gsrc, uint32_t lds_byte,
                                                 uint32_t tile_d0, uint32_t tile_d1,
                                                 uint32_t row_stride_elems) {
    const uint64_t ga = (uint64_t)(size_t)gsrc;     // global byte address
    tv4u g0 = {
        1u,                                          // count=1, user descriptor
        lds_byte,                                    // lds_addr
        (uint32_t)(ga & 0xFFFFFFFFu),                // global_addr[31:0]
        (uint32_t)((ga >> 32) & 0x01FFFFFFu) | (2u << 30)   // addr[56:32] | type=2
    };
    tv8i g1 = {
        (int)((1u << 16) |                           // data_size = 2 bytes
              (1u << 20) |                           // pad_enable
              (3u << 22) |                           // pad_interval: 16 DWORDs
              (3u << 25)),                           // pad_amount:    4 DWORDs
        (int)((row_stride_elems & 0xFFFFu) << 16),   // tensor_dim0[15:0] @ 63:48
        (int)(((row_stride_elems >> 16) & 0xFFFFu) | // tensor_dim0[31:16]
              ((tile_d1 & 0xFFFFu) << 16)),          // tensor_dim1[15:0] @ 95:80
        (int)((tile_d0 & 0xFFFFu) << 16),            // tile_dim0 @ 127:112
        (int)(tile_d1 & 0xFFFFu),                    // tile_dim1 @ 143:128
        (int)row_stride_elems,                       // tensor_dim0_stride[31:0]
        0, 0
    };
    tv4i gz = {0, 0, 0, 0};
#if defined(__clang_major__) && (__clang_major__ >= 23)
    tv8i gz8 = {0, 0, 0, 0, 0, 0, 0, 0};
    __builtin_amdgcn_tensor_load_to_lds(g0, g1, gz, gz, gz8, 0);
#else
    __builtin_amdgcn_tensor_load_to_lds(g0, g1, gz, gz, 0);
#endif
}

__device__ __forceinline__ uint32_t lds_byte_offset(const void* p) {
    return (uint32_t)(size_t)(__attribute__((address_space(3))) const void*)p;
}
#endif

// A fragment: 16x32 bf16, element (M,K); base points at (m0,k0) of a
// row-major (stride in elements) matrix.  ISA 7.12.2 layout:
//   lane L: M = L&15 ; K chunks [kh..kh+7] and [kh+16..kh+23], kh=(L>>4)*8
__device__ __forceinline__ v16bf load_frag_a(const u16* base, int stride, int lane) {
    const u16* p = base + (size_t)(lane & 15) * stride + ((lane >> 4) * 8);
    FragU f;
    f.q[0] = *(const uint4*)(p);
    f.q[1] = *(const uint4*)(p + 16);
    return f.v;
}

// B fragment: 32x16 bf16, element (K,N); storage is "transposed": element
// (K,N) lives at baseT[n*stride + K]  (contiguous over K).
//   lane L: N = L&15 ; K chunks [kh..kh+7] and [kh+16..kh+23], kh=(L>>4)*8
__device__ __forceinline__ v16bf load_frag_b(const u16* baseT, size_t stride, int lane) {
    const u16* p = baseT + (size_t)(lane & 15) * stride + ((lane >> 4) * 8);
    FragU f;
    f.q[0] = *(const uint4*)(p);
    f.q[1] = *(const uint4*)(p + 16);
    return f.v;
}

__device__ __forceinline__ v8f wmma_bf16(v16bf a, v16bf b, v8f c) {
    return __builtin_amdgcn_wmma_f32_16x16x32_bf16(false, a, false, b,
                                                   (short)0, c, false, false);
}

__device__ __forceinline__ float redmax16(float v) {
    #pragma unroll
    for (int m = 1; m < 16; m <<= 1) v = fmaxf(v, __shfl_xor(v, m, 32));
    return v;
}
__device__ __forceinline__ float redsum16(float v) {
    #pragma unroll
    for (int m = 1; m < 16; m <<= 1) v += __shfl_xor(v, m, 32);
    return v;
}

// ------------------------------------------------------- convert kernels

// x (B, C, S) f32  ->  xT (B, S, C) bf16
__global__ __launch_bounds__(256) void transpose_x_kernel(const float* __restrict__ x,
                                                          u16* __restrict__ xT) {
    __shared__ float tile[32][33];
    int s0 = blockIdx.x * 32, c0 = blockIdx.y * 32, b = blockIdx.z;
    int tx = threadIdx.x, ty = threadIdx.y;   // block (32,8)
    #pragma unroll
    for (int i = 0; i < 4; ++i) {
        int c = c0 + ty + i * 8;
        tile[ty + i * 8][tx] = x[((size_t)b * Cc + c) * Sn + s0 + tx];
    }
    __syncthreads();
    #pragma unroll
    for (int i = 0; i < 4; ++i) {
        int s = s0 + ty + i * 8;
        xT[((size_t)b * Sn + s) * Cc + c0 + tx] = f2bf_bits(tile[tx][ty + i * 8]);
    }
}

// generic f32 -> bf16, 4 elements per thread
__global__ __launch_bounds__(256) void cvt_bf16_kernel(const float* __restrict__ src,
                                                       u16* __restrict__ dst, int n4) {
    int i = blockIdx.x * blockDim.x + threadIdx.x;
    if (i >= n4) return;
    float4 f = ((const float4*)src)[i];
    union { u16 u[4]; uint2 v; } o;
    o.u[0] = f2bf_bits(f.x); o.u[1] = f2bf_bits(f.y);
    o.u[2] = f2bf_bits(f.z); o.u[3] = f2bf_bits(f.w);
    ((uint2*)dst)[i] = o.v;
}

// ------------------------------------------------------- projection GEMM
// Y[b](o,s) = sum_c W(o,c) * X[b](c,s) + bias(o)
// Block tile 128x128, 8 waves of 32x64, K-step 32.
// A (128x32) and B (128x32, stored [n][k]) tiles double-buffered in LDS,
// filled with global_load_async_to_lds_b128 (ASYNCcnt).
// MODE 0: q -> bf16(B,O,S)                   MODE 1: k -> bf16 + f32 d_out
// MODE 2: v -> f32 d_out + bf16 vT(B,H,S,D)  MODE 3: out -> f32 d_out
template <int MODE>
__global__ __launch_bounds__(256) void proj_kernel(const u16* __restrict__ Wb,
                                                   const float* __restrict__ bias,
                                                   const u16* __restrict__ Bt,
                                                   u16* __restrict__ outb,
                                                   float* __restrict__ outf,
                                                   u16* __restrict__ vT) {
    __shared__ u16 a_lds[2][128][40];   // [buf][m][k], padded rows (80B)
    __shared__ u16 b_lds[2][128][40];   // [buf][n][k]

    const int tid  = threadIdx.x;
    const int lane = tid & 31;
    const int wave = tid >> 5;
    const int b     = blockIdx.z;
    const int m_blk = blockIdx.y * 128;
    const int n_blk = blockIdx.x * 128;
    const int m0l   = (wave & 3) * 32;   // local M strip (2 tiles)
    const int n0l   = (wave >> 2) * 64;  // local N strip (4 tiles)

    // cooperative-staging coordinates: 512 16B-chunks per tile, 2 per thread
    const int rowA0 = tid >> 1,          cA0 = (tid & 1) * 2;       // chunks c,c+1
    const u16* gA_row = Wb + (size_t)(m_blk + rowA0) * Cc + cA0 * 8;
    const u16* gB_row = Bt + ((size_t)b * Sn + n_blk + rowA0) * Cc + cA0 * 8;

#if USE_ASYNC_LDS
    auto stage = [&](int buf, int k0) {
        const u16* ga = gA_row + k0;
        const u16* gb = gB_row + k0;
        u16* la = &a_lds[buf][rowA0][cA0 * 8];
        u16* lb = &b_lds[buf][rowA0][cA0 * 8];
        async_ld16(ga,     la);
        async_ld16(ga + 8, la + 8);
        async_ld16(gb,     lb);
        async_ld16(gb + 8, lb + 8);
    };
#else
    uint4 ra0, ra1, rb0, rb1;
    auto fetch = [&](int k0) {
        ra0 = *(const uint4*)(gA_row + k0);
        ra1 = *(const uint4*)(gA_row + k0 + 8);
        rb0 = *(const uint4*)(gB_row + k0);
        rb1 = *(const uint4*)(gB_row + k0 + 8);
    };
    auto commit = [&](int buf) {
        *(uint4*)&a_lds[buf][rowA0][cA0 * 8]     = ra0;
        *(uint4*)&a_lds[buf][rowA0][cA0 * 8 + 8] = ra1;
        *(uint4*)&b_lds[buf][rowA0][cA0 * 8]     = rb0;
        *(uint4*)&b_lds[buf][rowA0][cA0 * 8 + 8] = rb1;
    };
#endif

    v8f acc[2][4];
    #pragma unroll
    for (int i = 0; i < 2; ++i)
        #pragma unroll
        for (int j = 0; j < 4; ++j) acc[i][j] = (v8f)0.0f;

    auto compute = [&](int buf) {
        v16bf a0 = load_frag_a(&a_lds[buf][m0l][0],      40, lane);
        v16bf a1 = load_frag_a(&a_lds[buf][m0l + 16][0], 40, lane);
        #pragma unroll
        for (int j = 0; j < 4; ++j) {
            v16bf bf = load_frag_b(&b_lds[buf][n0l + j * 16][0], 40, lane);
            acc[0][j] = wmma_bf16(a0, bf, acc[0][j]);
            acc[1][j] = wmma_bf16(a1, bf, acc[1][j]);
        }
    };

    constexpr int KSTEPS = Cc / 32;   // 128

#if USE_ASYNC_LDS
    stage(0, 0);
    for (int kb = 0; kb < KSTEPS - 1; ++kb) {
        stage((kb + 1) & 1, (kb + 1) * 32);       // prefetch next tile
        __builtin_amdgcn_s_wait_asynccnt(4);      // current tile landed
        __syncthreads();
        compute(kb & 1);
        __syncthreads();                           // done reading before overwrite
    }
    __builtin_amdgcn_s_wait_asynccnt(0);
    __syncthreads();
    compute((KSTEPS - 1) & 1);
#else
    fetch(0);
    commit(0);
    for (int kb = 0; kb < KSTEPS - 1; ++kb) {
        fetch((kb + 1) * 32);                      // issue loads early
        __syncthreads();
        compute(kb & 1);                           // overlap with in-flight loads
        __syncthreads();
        commit((kb + 1) & 1);
    }
    __syncthreads();
    compute((KSTEPS - 1) & 1);
#endif

    // -------- epilogue
    const int half = lane >> 4, col = lane & 15;
    #pragma unroll
    for (int i = 0; i < 2; ++i) {
        const int mt = m_blk + m0l + i * 16;
        const int mrow0 = mt + 8 * half;
        union { float f[8]; float4 v4[2]; } bb;
        bb.v4[0] = *(const float4*)(bias + mrow0);
        bb.v4[1] = *(const float4*)(bias + mrow0 + 4);
        #pragma unroll
        for (int j = 0; j < 4; ++j) {
            const int n = n_blk + n0l + j * 16 + col;
            float vals[8];
            #pragma unroll
            for (int r = 0; r < 8; ++r) vals[r] = acc[i][j][r] + bb.f[r];

            if (MODE == 0 || MODE == 1) {
                #pragma unroll
                for (int r = 0; r < 8; ++r)
                    outb[((size_t)b * Cc + mrow0 + r) * Sn + n] = f2bf_bits(vals[r]);
            }
            if (MODE == 1 || MODE == 2 || MODE == 3) {
                #pragma unroll
                for (int r = 0; r < 8; ++r)
                    outf[((size_t)b * Cc + mrow0 + r) * Sn + n] = vals[r];
            }
            if (MODE == 2) {
                Pack8 vtp;
                #pragma unroll
                for (int r = 0; r < 8; ++r) vtp.u[r] = f2bf_bits(vals[r]);
                const int hh = mt >> 7;               // head index
                const int d0 = (mt & 127) + 8 * half; // head-dim base (8-aligned)
                *(uint4*)(vT + (((size_t)(b * Hn + hh) * Sn + n) * Dn + d0)) = vtp.q;
            }
        }
    }
}

// ------------------------------------------------------- fused attention
// Per (b,h): scores = q k^T / sqrt(S), causal over (D,D), softmax, p @ v,
// write attnT (B, S, C) bf16 for the output projection.
// k tiles staged into LDS via the Tensor Data Mover (double-buffered,
// wave 0 issues, TENSORcnt + barrier sync); q / v pipelined in registers.
__global__ __launch_bounds__(256) void attention_kernel(const u16* __restrict__ qb,
                                                        const u16* __restrict__ kb,
                                                        const u16* __restrict__ vT,
                                                        u16* __restrict__ attnT) {
    __shared__ u16 k_lds[2][128][40];   // [buf][e][k] 32-elem rows + 4-DW pad
    __shared__ u16 p_lds[8][16][136];   // per-wave 16x128 p tile (bf16), padded

    const int tid  = threadIdx.x;
    const int lane = tid & 31;
    const int wave = tid >> 5;
    const int b = blockIdx.x >> 5;
    const int h = blockIdx.x & 31;
    const int m0 = wave * 16;           // 16-row strip of d owned by this wave

    // ---- phase 1: scores (128x128), K over S
    const u16* qbase = qb + ((size_t)b * Cc + h * Dn + m0) * Sn;
    const u16* kbase = kb + ((size_t)b * Cc + h * Dn) * Sn;

    v8f sc[8];
    #pragma unroll
    for (int t = 0; t < 8; ++t) sc[t] = (v8f)0.0f;

    constexpr int KB1 = Sn / 32;        // 64 k-steps

#if USE_TDM
    {
        const uint32_t lds0 = lds_byte_offset(&k_lds[0][0][0]);
        const uint32_t lds1 = lds_byte_offset(&k_lds[1][0][0]);
        if (wave == 0) tdm_load_tile_2d(kbase, lds0, 32, 128, Sn);
        v16bf a_cur = load_frag_a(qbase, Sn, lane);
        for (int kblk = 0; kblk < KB1 - 1; ++kblk) {
            const int k0n = (kblk + 1) * 32;
            if (wave == 0)
                tdm_load_tile_2d(kbase + k0n, ((kblk + 1) & 1) ? lds1 : lds0,
                                 32, 128, Sn);
            v16bf a_nxt = load_frag_a(qbase + k0n, Sn, lane);   // overlap
            if (wave == 0) __builtin_amdgcn_s_wait_tensorcnt(1);
            __syncthreads();
            #pragma unroll
            for (int t = 0; t < 8; ++t) {
                v16bf bf = load_frag_b(&k_lds[kblk & 1][t * 16][0], 40, lane);
                sc[t] = wmma_bf16(a_cur, bf, sc[t]);
            }
            __syncthreads();
            a_cur = a_nxt;
        }
        if (wave == 0) __builtin_amdgcn_s_wait_tensorcnt(0);
        __syncthreads();
        #pragma unroll
        for (int t = 0; t < 8; ++t) {
            v16bf bf = load_frag_b(&k_lds[(KB1 - 1) & 1][t * 16][0], 40, lane);
            sc[t] = wmma_bf16(a_cur, bf, sc[t]);
        }
    }
#else
    for (int kblk = 0; kblk < KB1; ++kblk) {
        const int k0 = kblk * 32;
        v16bf a = load_frag_a(qbase + k0, Sn, lane);
        #pragma unroll
        for (int t = 0; t < 8; ++t) {
            v16bf bf = load_frag_b(kbase + (size_t)(t * 16) * Sn + k0, Sn, lane);
            sc[t] = wmma_bf16(a, bf, sc[t]);
        }
    }
#endif

    // ---- masked softmax over e (row-wise), all in registers + shuffles
    const float scale = 0.022097086912079608f;   // 1/sqrt(2048)
    const int half = lane >> 4, ncol = lane & 15;
    #pragma unroll
    for (int r = 0; r < 8; ++r) {
        const int d = m0 + 8 * half + r;
        float mx = -3.402823466e38f;
        #pragma unroll
        for (int t = 0; t < 8; ++t) {
            float v = sc[t][r] * scale;
            const int e = t * 16 + ncol;
            v = (e <= d) ? v : -3.402823466e38f;
            sc[t][r] = v;
            mx = fmaxf(mx, v);
        }
        mx = redmax16(mx);
        float sum = 0.0f;
        #pragma unroll
        for (int t = 0; t < 8; ++t) {
            float ev = __expf(sc[t][r] - mx);
            sc[t][r] = ev;
            sum += ev;
        }
        sum = redsum16(sum);
        const float inv = 1.0f / sum;
        #pragma unroll
        for (int t = 0; t < 8; ++t)
            p_lds[wave][8 * half + r][t * 16 + ncol] = f2bf_bits(sc[t][r] * inv);
    }
    __syncthreads();

    // ---- phase 2: attn = p(16x128 strip) @ v(128 x S)
    const u16* vbase = vT + (size_t)(b * Hn + h) * Sn * Dn;  // element(K=e,N=s) at [s*128+e]
    const u16* pbase = &p_lds[wave][0][0];

    // p A-fragments are n-invariant: hoist (4 frags, from LDS)
    v16bf pf[4];
    #pragma unroll
    for (int kb2 = 0; kb2 < 4; ++kb2)
        pf[kb2] = load_frag_a(pbase + kb2 * 32, 136, lane);

    auto load_v = [&](v16bf* f, int nt) {
        #pragma unroll
        for (int kb2 = 0; kb2 < 4; ++kb2)
            f[kb2] = load_frag_b(vbase + (size_t)(nt * 16) * Dn + kb2 * 32, Dn, lane);
    };
    auto compute_store = [&](const v16bf* f, int nt) {
        v8f acc = (v8f)0.0f;
        #pragma unroll
        for (int kb2 = 0; kb2 < 4; ++kb2)
            acc = wmma_bf16(pf[kb2], f[kb2], acc);
        Pack8 st;
        #pragma unroll
        for (int r = 0; r < 8; ++r) st.u[r] = f2bf_bits(acc[r]);
        const int s = nt * 16 + ncol;
        *(uint4*)(attnT + ((size_t)b * Sn + s) * Cc + h * Dn + m0 + 8 * half) = st.q;
    };

    v16bf vc[4], vn[4];
    load_v(vc, 0);
    constexpr int NT = Sn / 16;          // 128 (even)
    for (int nt = 0; nt < NT; nt += 2) {
        load_v(vn, nt + 1);              // prefetch while computing nt
        compute_store(vc, nt);
        if (nt + 2 < NT) load_v(vc, nt + 2);
        compute_store(vn, nt + 1);
    }
}

// ---------------------------------------------------------------- launch

extern "C" void kernel_launch(void* const* d_in, const int* in_sizes, int n_in,
                              void* d_out, int out_size, void* d_ws, size_t ws_size,
                              hipStream_t stream) {
    (void)in_sizes; (void)n_in; (void)out_size; (void)ws_size;

    const float* x  = (const float*)d_in[0];
    const float* Wq = (const float*)d_in[1];
    const float* bq = (const float*)d_in[2];
    const float* Wk = (const float*)d_in[3];
    const float* bk = (const float*)d_in[4];
    const float* Wv = (const float*)d_in[5];
    const float* bv = (const float*)d_in[6];
    const float* Wo = (const float*)d_in[7];
    const float* bo = (const float*)d_in[8];

    float* out = (float*)d_out;
    const size_t ELT = (size_t)Bn * Cc * Sn;          // 33,554,432
    float* kout = out + ELT;
    float* vout = out + 2 * ELT;

    // workspace layout (bytes)
    char* ws = (char*)d_ws;
    const size_t XT_B  = ELT * sizeof(u16);           // 64 MB, also reused as attnT
    const size_t W_B   = (size_t)Cc * Cc * sizeof(u16);
    u16* xT    = (u16*)(ws);
    u16* attnT = (u16*)(ws);                          // reuses xT after q/k/v proj
    u16* Wqb   = (u16*)(ws + XT_B);
    u16* Wkb   = (u16*)(ws + XT_B + W_B);
    u16* Wvb   = (u16*)(ws + XT_B + 2 * W_B);
    u16* Wob   = (u16*)(ws + XT_B + 3 * W_B);
    u16* qbuf  = (u16*)(ws + XT_B + 4 * W_B);
    u16* kbuf  = (u16*)(ws + XT_B + 4 * W_B + XT_B);
    u16* vTbuf = (u16*)(ws + XT_B + 4 * W_B + 2 * XT_B);

    // 1) x -> xT bf16 (transpose)
    transpose_x_kernel<<<dim3(Sn / 32, Cc / 32, Bn), dim3(32, 8), 0, stream>>>(x, xT);

    // 2) weights -> bf16
    {
        const int n4 = Cc * Cc / 4;
        const int blk = (n4 + 255) / 256;
        cvt_bf16_kernel<<<blk, 256, 0, stream>>>(Wq, Wqb, n4);
        cvt_bf16_kernel<<<blk, 256, 0, stream>>>(Wk, Wkb, n4);
        cvt_bf16_kernel<<<blk, 256, 0, stream>>>(Wv, Wvb, n4);
        cvt_bf16_kernel<<<blk, 256, 0, stream>>>(Wo, Wob, n4);
    }

    const dim3 pg(Sn / 128, Cc / 128, Bn);
    // 3) q / k / v projections (WMMA, async-LDS double-buffered)
    proj_kernel<0><<<pg, 256, 0, stream>>>(Wqb, bq, xT, qbuf, nullptr, nullptr);
    proj_kernel<1><<<pg, 256, 0, stream>>>(Wkb, bk, xT, kbuf, kout, nullptr);
    proj_kernel<2><<<pg, 256, 0, stream>>>(Wvb, bv, xT, nullptr, vout, vTbuf);

    // 4) fused attention per (b,h), writes attnT (reuses xT region)
    attention_kernel<<<Bn * Hn, 256, 0, stream>>>(qbuf, kbuf, vTbuf, attnT);

    // 5) output projection -> d_out
    proj_kernel<3><<<pg, 256, 0, stream>>>(Wob, bo, attnT, nullptr, out, nullptr);
}
